// NLSA_55164559949868
// MI455X (gfx1250) — matrix operand order repaired and checked by reading, not compiled, not run on hardware
//
#include <hip/hip_runtime.h>

typedef __attribute__((ext_vector_type(8)))  __bf16 v8bf;
typedef __attribute__((ext_vector_type(16))) __bf16 v16bf;
typedef __attribute__((ext_vector_type(8)))  float  v8f;

#define NB_   8      // batch N
#define CIN   256
#define HW_   9216
#define NATT  4
#define KB_   144    // bucket size
#define NBUCK 64
#define COUT  256
#define MPROJ 64
#define NPAD  16384
#define XROW  264    // padded LDS row stride in bf16 (528B -> 4-bank rotation/row)
#define M1STR 148    // padded column stride (floats) for the m1 staging buffer

// ---------------- k0: NCHW f32 -> (N, HW, C) bf16 tiled transpose ----------------
__global__ void k0_transpose(const float* __restrict__ in, __bf16* __restrict__ xT) {
  __shared__ float tile[32][33];
  int n = blockIdx.z;
  int c0 = blockIdx.y * 32;
  int s0 = blockIdx.x * 32;
  int tx = threadIdx.x, ty = threadIdx.y;
#pragma unroll
  for (int r = 0; r < 4; ++r) {
    int c = c0 + ty + 8 * r;
    tile[ty + 8 * r][tx] = in[((size_t)(n * CIN + c)) * HW_ + s0 + tx];
  }
  __syncthreads();
#pragma unroll
  for (int r = 0; r < 4; ++r) {
    int s = s0 + ty + 8 * r;
    xT[((size_t)n * HW_ + s) * CIN + c0 + tx] = (__bf16)tile[tx][ty + 8 * r];
  }
}

// ---------------- k1: hash = max_m( x[n,s,:] @ R_a[:,m] ) ----------------
__global__ void k1_hash(const float* __restrict__ in, const float* __restrict__ rm,
                        float* __restrict__ hash) {
  extern __shared__ char _sm1[];
  float* R = (float*)_sm1;                       // 256*64 f32 = 64KB
  int tid = threadIdx.x;
  int na = blockIdx.y;                           // na = n*4 + a
  int a = na & 3, n = na >> 2;
  int s = blockIdx.x * 128 + tid;
  for (int i = tid; i < CIN * MPROJ; i += 128) R[i] = rm[a * CIN * MPROJ + i];
  __syncthreads();
  float acc[MPROJ];
#pragma unroll
  for (int m = 0; m < MPROJ; ++m) acc[m] = 0.f;
#pragma unroll 1
  for (int c = 0; c < CIN; ++c) {
    float xv = in[((size_t)(n * CIN + c)) * HW_ + s];   // coalesced over s
#pragma unroll
    for (int m = 0; m < MPROJ; ++m) acc[m] = fmaf(xv, R[c * MPROJ + m], acc[m]);
  }
  float h = acc[0];
#pragma unroll
  for (int m = 1; m < MPROJ; ++m) h = fmaxf(h, acc[m]);
  hash[(size_t)na * HW_ + s] = h;
}

// ---------------- k2: per-(n,a) bitonic argsort, descending ----------------
__global__ void k2_sort(const float* __restrict__ hash, int* __restrict__ sidx) {
  extern __shared__ char _sm2[];
  float* key = (float*)_sm2;                     // 16384 f32
  int* val = (int*)(_sm2 + NPAD * 4);            // 16384 i32
  int tid = threadIdx.x;
  int row = blockIdx.x;
  const float* hrow = hash + (size_t)row * HW_;
  for (int i = tid; i < NPAD; i += 1024) {
    key[i] = (i < HW_) ? hrow[i] : -3.4e38f;     // pad sinks to the end (descending)
    val[i] = i;
  }
  __syncthreads();
  for (int k = 2; k <= NPAD; k <<= 1) {
    for (int j = k >> 1; j > 0; j >>= 1) {
      for (int i = tid; i < NPAD; i += 1024) {
        int ixj = i ^ j;
        if (ixj > i) {
          float ki = key[i], kj = key[ixj];
          bool descBlock = ((i & k) == 0);
          bool sw = descBlock ? (ki < kj) : (ki > kj);
          if (sw) {
            key[i] = kj; key[ixj] = ki;
            int t = val[i]; val[i] = val[ixj]; val[ixj] = t;
          }
        }
      }
      __syncthreads();
    }
  }
  for (int i = tid; i < HW_; i += 1024) sidx[(size_t)row * HW_ + i] = val[i];
}

// ---------------- k3 helper: one column-block GEMM (9 M-tiles x K=256) ----------------
// A (16x32 bf16) per ISA 7.12.2: half 0 -> K = e / 16+e ; half 1 -> +8
// B (32x16 bf16): same K order as A: elems 0-7 K=8*half+0..7, elems 8-15 K=16+8*half+0..7
__device__ __forceinline__ void gemm_col(const __bf16* __restrict__ Xb,
                                         const float* __restrict__ W,
                                         int orow, int lcol, int half, v8f acc[9]) {
#pragma unroll 1
  for (int kc = 0; kc < 8; ++kc) {
    int K0 = kc * 32;
    const float* wr = W + (size_t)orow * CIN + K0 + half * 8;
    float4 f0 = *(const float4*)(wr);
    float4 f1 = *(const float4*)(wr + 4);
    float4 f2 = *(const float4*)(wr + 16);
    float4 f3 = *(const float4*)(wr + 20);
    v16bf B;
    B[0]  = (__bf16)f0.x; B[1]  = (__bf16)f0.y; B[2]  = (__bf16)f0.z; B[3]  = (__bf16)f0.w;
    B[4]  = (__bf16)f1.x; B[5]  = (__bf16)f1.y; B[6]  = (__bf16)f1.z; B[7]  = (__bf16)f1.w;
    B[8]  = (__bf16)f2.x; B[9]  = (__bf16)f2.y; B[10] = (__bf16)f2.z; B[11] = (__bf16)f2.w;
    B[12] = (__bf16)f3.x; B[13] = (__bf16)f3.y; B[14] = (__bf16)f3.z; B[15] = (__bf16)f3.w;
#pragma unroll
    for (int mt = 0; mt < 9; ++mt) {
      const __bf16* ar = Xb + (mt * 16 + lcol) * XROW + K0 + half * 8;
      v8bf alo = *(const v8bf*)ar;          // K = base+0..7
      v8bf ahi = *(const v8bf*)(ar + 16);   // K = base+16..23
      v16bf A = __builtin_shufflevector(alo, ahi, 0, 1, 2, 3, 4, 5, 6, 7,
                                        8, 9, 10, 11, 12, 13, 14, 15);
      acc[mt] = __builtin_amdgcn_wmma_f32_16x16x32_bf16(
          false, A, false, B, (short)0, acc[mt], false, false);
    }
  }
}

// ---------------- k3: async gather + dual GEMM + bucket attention + scatter ----------------
__global__ void __launch_bounds__(256)
k3_main(const __bf16* __restrict__ xT, const int* __restrict__ sidx,
        const float* __restrict__ tw, const float* __restrict__ tb,
        const float* __restrict__ gw, const float* __restrict__ gb,
        float* __restrict__ out) {
  extern __shared__ char _sm3[];
  int* sIdxLds = (int*)_sm3;                         // 144 ints (640B reserved)
  __bf16* Xb = (__bf16*)(_sm3 + 640);                // 144 x XROW bf16 = 76,032B
  float* M1c = (float*)(_sm3 + 640 + KB_ * XROW * 2);// col-major [256][M1STR] f32
  int tid = threadIdx.x;
  int bid = blockIdx.x;
  int b = bid & 63;
  int a = (bid >> 6) & 3;
  int n = bid >> 8;

  for (int i = tid; i < KB_; i += 256)
    sIdxLds[i] = sidx[((size_t)(n * NATT + a)) * HW_ + b * KB_ + i];
  __syncthreads();

  // async-gather 144 token rows (512B each) into LDS; tracked by ASYNCcnt
  for (int i = tid; i < KB_ * 32; i += 256) {        // 4608 = 18 * 256, all lanes active
    int t = i >> 5, ch = i & 31;
    const char* src =
        (const char*)(xT + ((size_t)n * HW_ + sIdxLds[t]) * CIN) + ch * 16;
    unsigned dstOff = (unsigned)(size_t)(Xb + t * XROW + ch * 8);  // LDS byte offset
    asm volatile("global_load_async_to_lds_b128 %0, %1, off"
                 :: "v"(dstOff), "v"(src) : "memory");
  }
  asm volatile("s_wait_asynccnt 0" ::: "memory");
  __syncthreads();

  int wave = tid >> 5, lane = tid & 31;
  int half = lane >> 4, lcol = lane & 15;
  float* outBase = out + ((size_t)(a * NB_ + n)) * HW_ * COUT;
  const v8f vzero = {0.f, 0.f, 0.f, 0.f, 0.f, 0.f, 0.f, 0.f};

#pragma unroll 1
  for (int otl = 0; otl < 2; ++otl) {
    int ot = wave * 2 + otl;                         // 16 o-tiles over 8 waves x 2
    int o = ot * 16 + lcol;
    float* mcol = M1c + (size_t)o * M1STR;           // wave-private m1 column
    v8f acc[9];

    // ---- m1 = G @ theta_w.T + theta_b : stage to LDS, fold partial sum ----
#pragma unroll
    for (int mt = 0; mt < 9; ++mt) acc[mt] = vzero;
    gemm_col(Xb, tw, o, lcol, half, acc);
    float biasT = tb[o];
    float sT = 0.f;
#pragma unroll
    for (int mt = 0; mt < 9; ++mt) {
      acc[mt] += biasT;
      int base = mt * 16 + 8 * half;
      float4 lo = {acc[mt][0], acc[mt][1], acc[mt][2], acc[mt][3]};
      float4 hi = {acc[mt][4], acc[mt][5], acc[mt][6], acc[mt][7]};
      sT += acc[mt][0] + acc[mt][1] + acc[mt][2] + acc[mt][3] +
            acc[mt][4] + acc[mt][5] + acc[mt][6] + acc[mt][7];
      *(float4*)(mcol + base) = lo;
      *(float4*)(mcol + base + 4) = hi;
    }

    // ---- m2 = G @ g_w.T + g_b : reuse acc, fold sum(m1*m2) from LDS ----
#pragma unroll
    for (int mt = 0; mt < 9; ++mt) acc[mt] = vzero;
    gemm_col(Xb, gw, o, lcol, half, acc);
    float biasG = gb[o];
    float sTG = 0.f;
#pragma unroll
    for (int mt = 0; mt < 9; ++mt) {
      acc[mt] += biasG;
      int base = mt * 16 + 8 * half;
      float4 lo = *(const float4*)(mcol + base);
      float4 hi = *(const float4*)(mcol + base + 4);
      sTG = fmaf(lo.x, acc[mt][0], sTG);
      sTG = fmaf(lo.y, acc[mt][1], sTG);
      sTG = fmaf(lo.z, acc[mt][2], sTG);
      sTG = fmaf(lo.w, acc[mt][3], sTG);
      sTG = fmaf(hi.x, acc[mt][4], sTG);
      sTG = fmaf(hi.y, acc[mt][5], sTG);
      sTG = fmaf(hi.z, acc[mt][6], sTG);
      sTG = fmaf(hi.w, acc[mt][7], sTG);
    }
    sT += __shfl_xor(sT, 16, 32);                    // combine the two lane halves
    sTG += __shfl_xor(sTG, 16, 32);

    // ---- att = (m1 * sTG) / (m1 * sT), scattered to original positions ----
#pragma unroll
    for (int mt = 0; mt < 9; ++mt) {
      int base = mt * 16 + 8 * half;
      float4 lo = *(const float4*)(mcol + base);
      float4 hi = *(const float4*)(mcol + base + 4);
      float m1s[8] = {lo.x, lo.y, lo.z, lo.w, hi.x, hi.y, hi.z, hi.w};
#pragma unroll
      for (int r = 0; r < 8; ++r) {
        float m1v = m1s[r];
        float att = (m1v * sTG) / (m1v * sT);
        outBase[(size_t)sIdxLds[base + r] * COUT + o] = att;
      }
    }
  }
}

// ---------------- host ----------------
extern "C" void kernel_launch(void* const* d_in, const int* in_sizes, int n_in,
                              void* d_out, int out_size, void* d_ws, size_t ws_size,
                              hipStream_t stream) {
  const float* inputs  = (const float*)d_in[0];  // (8,256,96,96)
  const float* rm      = (const float*)d_in[1];  // (4,256,64)
  const float* theta_w = (const float*)d_in[2];  // (256,256)
  const float* theta_b = (const float*)d_in[3];  // (256,)
  const float* g_w     = (const float*)d_in[4];  // (256,256)
  const float* g_b     = (const float*)d_in[5];  // (256,)
  float* out = (float*)d_out;

  char* ws = (char*)d_ws;
  __bf16* xT  = (__bf16*)ws;                               // 37,748,736 B
  float*  hsh = (float*)(ws + 37748736);                   //  1,179,648 B
  int*    sdx = (int*)(ws + 38928384);                     //  1,179,648 B

  k0_transpose<<<dim3(HW_ / 32, CIN / 32, NB_), dim3(32, 8), 0, stream>>>(inputs, xT);
  k1_hash<<<dim3(HW_ / 128, NB_ * NATT), 128, CIN * MPROJ * 4, stream>>>(inputs, rm, hsh);
  k2_sort<<<NB_ * NATT, 1024, NPAD * 8, stream>>>(hsh, sdx);
  size_t smem3 = 640 + (size_t)KB_ * XROW * 2 + (size_t)COUT * M1STR * 4;  // 228,224B
  k3_main<<<NB_ * NATT * NBUCK, 256, smem3, stream>>>(
      xT, sdx, theta_w, theta_b, g_w, g_b, out);
}
